// TemporalAttention_80126909874765
// MI455X (gfx1250) — matrix operand order repaired and checked by reading, not compiled, not run on hardware
//
#include <hip/hip_runtime.h>
#include <hip/hip_bf16.h>

typedef __attribute__((ext_vector_type(16))) _Float16 v16h;
typedef __attribute__((ext_vector_type(8)))  float    v8f;

#define RSQ_BN 0.9999950000374997f   // 1/sqrt(1+1e-5)
#define RSQ_DKH 0.70710678118654752f // DKH^-0.5, DKH=2

// ---------- WMMA fragment helpers (16x16x32 f16, wave32) ----------
// A-matrix 16x32 f16: lane m=lane&15 holds row m; lanes>=16 K offset +16;
// pairs p -> K=2p.
__device__ inline v16h load_a_frag_g(const _Float16* __restrict__ A, int row0, int k0, int ld) {
  const int lane = threadIdx.x & 31;
  const _Float16* base = A + (size_t)(row0 + (lane & 15)) * ld + k0 + ((lane & 16) ? 16 : 0);
  v16h a;
#pragma unroll
  for (int p = 0; p < 8; ++p) {
    int k = 2 * p;
    a[2 * p]     = base[k];
    a[2 * p + 1] = base[k + 1];
  }
  return a;
}

// B-matrix 32x16 f16 from LDS stored column(t)-major: element B[k][t] at L[t*ld + k].
// lane n=lane&15 holds col n; lanes>=16 K offset +16; element e -> K=e.
__device__ inline v16h load_b_frag_lds(const _Float16* __restrict__ L, int n0, int k0, int ld) {
  const int lane = threadIdx.x & 31;
  const _Float16* base = L + (n0 + (lane & 15)) * ld + k0 + ((lane & 16) ? 16 : 0);
  v16h b;
#pragma unroll
  for (int e = 0; e < 16; ++e) b[e] = base[e];
  return b;
}

// ---------- Kernel 0: weight conversion to f16 ----------
__global__ void prep_kernel(const float* __restrict__ qkvw, const float* __restrict__ tcnw,
                            _Float16* __restrict__ qkv16, _Float16* __restrict__ w216) {
  int i = blockIdx.x * 256 + threadIdx.x;
  if (i < 4096)  qkv16[i] = (_Float16)qkvw[i];
  if (i < 27648) w216[i]  = (_Float16)tcnw[i];   // tcn_w (48,64,9,1) flat == [co][ci*9+kt]
}

// ---------- Kernel 1: fused attention branch (one block per (n,v)) ----------
__global__ __launch_bounds__(256) void attn_kernel(
    const float* __restrict__ x, const float* __restrict__ pe,
    const float* __restrict__ dbg, const float* __restrict__ dbb,
    const _Float16* __restrict__ qkv16, const float* __restrict__ qkvb,
    const float* __restrict__ attn_w, const float* __restrict__ attn_b,
    float* __restrict__ pre)
{
  const int bid = blockIdx.x;
  const int n = bid / 25, v = bid % 25;
  const int tid = threadIdx.x;
  const int LDB = 70;                       // padded t-major row stride (bank-friendly)
  __shared__ _Float16 lx[64 * 70];          // data_bn(x)  panel, [t][c]
  __shared__ _Float16 ly[64 * 70];          // (x+pe)      panel, [t][c]
  __shared__ float fo[48 * 64];             // compact q(0..15) k(16..31) v(32..47) rows, [o][t]
  __shared__ float po[32 * 64];             // pos-branch q(0..15) k(16..31)
  __shared__ float lao[16 * 64];            // per-head attention output, [c][t]
  __shared__ float resx[16 * 64];           // raw x channels 48..63 for residual

  // load + data_bn + pe, cvt f16; stash raw channels 48..63 for the residual
  for (int i = tid; i < 4096; i += 256) {
    int c = i >> 6, t = i & 63;
    float xv = x[((size_t)(n * 64 + c) * 64 + t) * 25 + v];
    float xn = xv * (dbg[c * 25 + v] * RSQ_BN) + dbb[c * 25 + v];
    float yv = xv + pe[(c * 64 + t) * 25 + v];
    lx[t * LDB + c] = (_Float16)xn;
    ly[t * LDB + c] = (_Float16)yv;
    if (c >= 48) resx[(c - 48) * 64 + t] = xv;
  }
  __syncthreads();

  const int w = tid >> 5, lane = tid & 31;
  const int col = lane & 15, rowoff = (lane & 16) ? 8 : 0;

  // qkv GEMM: 12 x-branch tiles (rows {0-15,16-31,48-63}) + 8 y-branch tiles (rows 0-31)
  for (int tile = w; tile < 20; tile += 8) {
    int isY = (tile >= 12);
    int tt  = isY ? tile - 12 : tile;
    int m = tt >> 2, nt = tt & 3;
    int arow = isY ? m * 16 : ((m == 2) ? 48 : m * 16);  // global qkv output row base
    v8f acc = {};
#pragma unroll
    for (int ks = 0; ks < 2; ++ks) {
      v16h a = load_a_frag_g(qkv16, arow, ks * 32, 64);
      v16h b = load_b_frag_lds(isY ? ly : lx, nt * 16, ks * 32, LDB);
      acc = __builtin_amdgcn_wmma_f32_16x16x32_f16(false, a, false, b, (short)0, acc, false, false);
    }
#pragma unroll
    for (int r = 0; r < 8; ++r) {
      int og = arow + r + rowoff;            // global row for bias / q-scale
      int t  = nt * 16 + col;
      float val = acc[r] + qkvb[og];
      if (og < 16) val *= RSQ_DKH;           // q rows get DKH^-0.5
      int cr = m * 16 + r + rowoff;          // compact row
      if (isY) po[cr * 64 + t] = val;
      else     fo[cr * 64 + t] = val;
    }
  }
  __syncthreads();

  // per-head attention: wave w == head h (NH=8, DKH=DVH=2)
  {
    const int h = w;
    const float* fq0 = &fo[(2 * h) * 64];      const float* fq1 = &fo[(2 * h + 1) * 64];
    const float* fk0 = &fo[(16 + 2 * h) * 64]; const float* fk1 = &fo[(17 + 2 * h) * 64];
    const float* fv0 = &fo[(32 + 2 * h) * 64]; const float* fv1 = &fo[(33 + 2 * h) * 64];
    const float* pq0 = &po[(2 * h) * 64];      const float* pq1 = &po[(2 * h + 1) * 64];
    const float* pk0 = &po[(16 + 2 * h) * 64]; const float* pk1 = &po[(17 + 2 * h) * 64];
    // mu_q = mean_t fq
    float s0 = fq0[lane] + fq0[lane + 32];
    float s1 = fq1[lane] + fq1[lane + 32];
#pragma unroll
    for (int off = 16; off; off >>= 1) { s0 += __shfl_xor(s0, off, 32); s1 += __shfl_xor(s1, off, 32); }
    const float mu0 = s0 * (1.f / 64.f), mu1 = s1 * (1.f / 64.f);
    const int sa = lane, sb = lane + 32;       // two query positions per lane
    float fqa0 = fq0[sa] - mu0, fqa1 = fq1[sa] - mu1;
    float fqb0 = fq0[sb] - mu0, fqb1 = fq1[sb] - mu1;
    float pqa0 = pq0[sa], pqa1 = pq1[sa], pqb0 = pq0[sb], pqb1 = pq1[sb];
    float aoa0 = 0.f, aoa1 = 0.f, aob0 = 0.f, aob1 = 0.f;
#pragma unroll 4
    for (int t = 0; t < 64; ++t) {
      float k0v = fk0[t], k1v = fk1[t];
      float p0v = pk0[t], p1v = pk1[t];
      float v0v = fv0[t], v1v = fv1[t];
      float un = mu0 * k0v + mu1 * k1v;
      float lp = pqa0 * p0v + pqa1 * p1v;
      float pw = fqa0 * k0v + fqa1 * k1v;
      float wA = tanhf(pw + lp) + tanhf(un + lp);
      aoa0 += wA * v0v; aoa1 += wA * v1v;
      lp = pqb0 * p0v + pqb1 * p1v;
      pw = fqb0 * k0v + fqb1 * k1v;
      float wB = tanhf(pw + lp) + tanhf(un + lp);
      aob0 += wB * v0v; aob1 += wB * v1v;
    }
    lao[(2 * h) * 64 + sa] = aoa0; lao[(2 * h + 1) * 64 + sa] = aoa1;
    lao[(2 * h) * 64 + sb] = aob0; lao[(2 * h + 1) * 64 + sb] = aob1;
  }
  __syncthreads();

  // 16x16 attn projection + residual -> pre channels 48..63
  for (int i = tid; i < 1024; i += 256) {
    int o = i >> 6, s = i & 63;
    float acc = attn_b[o];
#pragma unroll
    for (int c = 0; c < 16; ++c) acc += attn_w[o * 16 + c] * lao[c * 64 + s];
    size_t gi = ((size_t)(n * 64 + 48 + o) * 64 + s) * 25 + v;
    pre[gi] = acc + resx[o * 64 + s];
  }
}

// ---------- Kernel 2: temporal conv via implicit GEMM (one block per (n,v)) ----------
__global__ __launch_bounds__(256) void tcn_kernel(
    const float* __restrict__ x, const _Float16* __restrict__ w216,
    const float* __restrict__ tb, const float* __restrict__ tg,
    const float* __restrict__ tbb, float* __restrict__ pre)
{
  const int bid = blockIdx.x;
  const int n = bid / 25, v = bid % 25;
  const int tid = threadIdx.x;
  __shared__ _Float16 xpad[64 * 72];        // [ci][t'], t' = t+4 halo (pad=4 both sides)
  __shared__ unsigned short off9[576];      // k -> (k/9)*72 + k%9  (im2col LDS offset)

  for (int i = tid; i < 64 * 72; i += 256) {
    int ci = i / 72, tp = i % 72;
    int t = tp - 4;
    float val = (t >= 0 && t < 64) ? x[((size_t)(n * 64 + ci) * 64 + t) * 25 + v] : 0.f;
    xpad[i] = (_Float16)val;
  }
  for (int k = tid; k < 576; k += 256) {
    int ci = k / 9;
    off9[k] = (unsigned short)(ci * 72 + (k - ci * 9));
  }
  __syncthreads();

  const int w = tid >> 5, lane = tid & 31;
  const int col = lane & 15, rowoff = (lane & 16) ? 8 : 0;
  const int koff = (lane & 16) ? 16 : 0;

  // GEMM: M=48 (co), N=64 (t), K=576 (ci*9+dt); 3x4 tiles, 18 k-steps each
  for (int tile = w; tile < 12; tile += 8) {
    int mt = tile >> 2, nt = tile & 3;
    int t0 = nt * 16 + col;
    v8f acc = {};
#pragma unroll 3
    for (int ks = 0; ks < 18; ++ks) {
      v16h a = load_a_frag_g(w216, mt * 16, ks * 32, 576);
      const unsigned short* tab = &off9[ks * 32 + koff];
      v16h b;
#pragma unroll
      for (int e = 0; e < 16; ++e)            // im2col gather via precomputed offsets
        b[e] = xpad[t0 + (int)tab[e]];
      acc = __builtin_amdgcn_wmma_f32_16x16x32_f16(false, a, false, b, (short)0, acc, false, false);
    }
#pragma unroll
    for (int r = 0; r < 8; ++r) {
      int o = mt * 16 + r + rowoff;           // out channel 0..47
      int t = nt * 16 + col;
      size_t gi = ((size_t)(n * 64 + o) * 64 + t) * 25 + v;
      pre[gi] = (acc[r] + tb[o]) * (tg[o] * RSQ_BN) + tbb[o] + x[gi];
    }
  }
}

// ---------- Kernel 3: SE gate (one block per n) ----------
__global__ __launch_bounds__(256) void se_kernel(
    const float* __restrict__ pre,
    const float* __restrict__ f1w, const float* __restrict__ f1b,
    const float* __restrict__ f2w, const float* __restrict__ f2b,
    float* __restrict__ gate)
{
  const int n = blockIdx.x;
  const int tid = threadIdx.x;
  __shared__ float part[256];
  __shared__ float sse[64];
  __shared__ float sh1[32];
  int c = tid >> 2, g = tid & 3;
  const float* basep = pre + (size_t)(n * 64 + c) * 1600;   // T*V = 1600
  float s = 0.f;
  for (int j = g; j < 1600; j += 4) s += basep[j];
  part[tid] = s;
  __syncthreads();
  if (g == 0) sse[c] = (part[tid] + part[tid + 1] + part[tid + 2] + part[tid + 3]) * (1.f / 1600.f);
  __syncthreads();
  if (tid < 32) {
    float a = f1b[tid];
#pragma unroll 8
    for (int cc = 0; cc < 64; ++cc) a += sse[cc] * f1w[tid * 64 + cc];
    sh1[tid] = a > 0.f ? a : 0.f;
  }
  __syncthreads();
  if (tid < 64) {
    float a = f2b[tid];
#pragma unroll 8
    for (int j = 0; j < 32; ++j) a += sh1[j] * f2w[tid * 32 + j];
    gate[n * 64 + tid] = 1.f / (1.f + __expf(-a));
  }
}

// ---------- Kernel 4: gating + final BN + relu, in place ----------
__global__ __launch_bounds__(256) void final_kernel(
    float* __restrict__ pre, const float* __restrict__ gate,
    const float* __restrict__ bng, const float* __restrict__ bnb)
{
  size_t idx = (size_t)blockIdx.x * 256 + threadIdx.x;
  if (idx >= 6553600ull) return;
  int c = (int)((idx / 1600) & 63);
  int n = (int)(idx / 102400);
  float val = pre[idx] * (1.f + gate[n * 64 + c]);
  float r = val * (bng[c] * RSQ_BN) + bnb[c];
  pre[idx] = r > 0.f ? r : 0.f;
}

extern "C" void kernel_launch(void* const* d_in, const int* in_sizes, int n_in,
                              void* d_out, int out_size, void* d_ws, size_t ws_size,
                              hipStream_t stream) {
  const float* x      = (const float*)d_in[0];
  const float* pe     = (const float*)d_in[1];
  const float* dbg    = (const float*)d_in[2];
  const float* dbb    = (const float*)d_in[3];
  const float* qkvw   = (const float*)d_in[4];
  const float* qkvb   = (const float*)d_in[5];
  const float* attn_w = (const float*)d_in[6];
  const float* attn_b = (const float*)d_in[7];
  const float* tcnw   = (const float*)d_in[8];
  const float* tcnb   = (const float*)d_in[9];
  const float* tcng   = (const float*)d_in[10];
  const float* tcnbb  = (const float*)d_in[11];
  const float* f1w    = (const float*)d_in[12];
  const float* f1b    = (const float*)d_in[13];
  const float* f2w    = (const float*)d_in[14];
  const float* f2b    = (const float*)d_in[15];
  const float* bng    = (const float*)d_in[16];
  const float* bnb    = (const float*)d_in[17];

  _Float16* qkv16 = (_Float16*)d_ws;                          // 4096 halfs
  _Float16* w216  = (_Float16*)((char*)d_ws + 8192);          // 27648 halfs
  float*    gate  = (float*)((char*)d_ws + 8192 + 55296);     // 64*64 floats
  float*    pre   = (float*)d_out;                            // (N,64,T,V) staging == output

  prep_kernel<<<108, 256, 0, stream>>>(qkvw, tcnw, qkv16, w216);
  attn_kernel<<<1600, 256, 0, stream>>>(x, pe, dbg, dbb, qkv16, qkvb, attn_w, attn_b, pre);
  tcn_kernel<<<1600, 256, 0, stream>>>(x, w216, tcnb, tcng, tcnbb, pre);
  se_kernel<<<64, 256, 0, stream>>>(pre, f1w, f1b, f2w, f2b, gate);
  final_kernel<<<25600, 256, 0, stream>>>(pre, gate, bng, bnb);
}
